// MoEFNN_10222022165090
// MI455X (gfx1250) — matrix operand order repaired and checked by reading, not compiled, not run on hardware
//
#include <hip/hip_runtime.h>
#include <hip/hip_bf16.h>
#include <math.h>

typedef __attribute__((ext_vector_type(16))) _Float16 v16h;
typedef __attribute__((ext_vector_type(8)))  _Float16 v8h;
typedef __attribute__((ext_vector_type(8)))  float    v8f;

#define D_IN     1024
#define D_HID    2048
#define NE       8
#define T_TOK    8192
#define CAP      T_TOK          // worst case: every token picks this expert
#define TILE_M   32             // 2 WMMA M-subtiles per block; B fragments reused 2x
#define MAX_TILES (T_TOK / TILE_M)   // 256 (counts[e] <= 8192)

// ---------------------------------------------------------------- zero init
__global__ __launch_bounds__(64) void zero_kernel(int* counts, float* importance) {
  int i = threadIdx.x;
  if (i < NE) { counts[i] = 0; importance[i] = 0.f; }
}

// ------------------------------------------- weight convert + transpose (f32 -> f16)
// W1  [E][D_IN][D_HID] -> W1T [E][D_HID][D_IN]   (K = D_IN contiguous per row)
// W2  [E][D_HID][D_IN] -> W2T [E][D_IN][D_HID]   (K = D_HID contiguous per row)
__global__ __launch_bounds__(256) void wprep_kernel(
    const float* __restrict__ W1, const float* __restrict__ W2,
    _Float16* __restrict__ W1T, _Float16* __restrict__ W2T) {
  size_t i = (size_t)blockIdx.x * 256 + threadIdx.x;     // < NE*D_IN*D_HID
  const size_t per = (size_t)D_IN * D_HID;
  int e = (int)(i / per);
  size_t r = i % per;
  int nh = (int)(r / D_IN);          // 0..2047
  int k  = (int)(r % D_IN);          // 0..1023
  W1T[i] = (_Float16)W1[(size_t)e * per + (size_t)k * D_HID + nh];
  int d = (int)(r / D_HID);          // 0..1023
  int h = (int)(r % D_HID);          // 0..2047
  W2T[i] = (_Float16)W2[(size_t)e * per + (size_t)h * D_IN + d];
}

// ---------------------------------------------------------------- gating (top-2)
__global__ __launch_bounds__(256) void gate_kernel(
    const float* __restrict__ x, const float* __restrict__ Wg,
    int* __restrict__ counts, float* __restrict__ importance,
    int* __restrict__ disp_entry, float* __restrict__ disp_gate) {
  int wave = threadIdx.x >> 5, lane = threadIdx.x & 31;
  int t = blockIdx.x * 8 + wave;

  float lacc[NE];
#pragma unroll
  for (int e = 0; e < NE; ++e) lacc[e] = 0.f;

  const float* xr = x + (size_t)t * D_IN;
  for (int d = lane; d < D_IN; d += 32) {
    float xv = xr[d];
    const float* wr = Wg + (size_t)d * NE;
#pragma unroll
    for (int e = 0; e < NE; ++e) lacc[e] += xv * wr[e];
  }
#pragma unroll
  for (int off = 16; off; off >>= 1) {
#pragma unroll
    for (int e = 0; e < NE; ++e) lacc[e] += __shfl_down(lacc[e], off, 32);
  }

  if (lane == 0) {
    // stable top-2 (lowest index wins ties, like lax.top_k)
    int i0 = 0; float v0 = lacc[0];
#pragma unroll
    for (int e = 1; e < NE; ++e) if (lacc[e] > v0) { v0 = lacc[e]; i0 = e; }
    int i1 = -1; float v1 = -3.4e38f;
#pragma unroll
    for (int e = 0; e < NE; ++e) if (e != i0 && lacc[e] > v1) { v1 = lacc[e]; i1 = e; }
    // softmax over [v0, v1] (v0 is the max)
    float ex = expf(v1 - v0);
    float den = 1.f + ex;
    float g0 = 1.f / den, g1 = ex / den;

    int s0 = atomicAdd(&counts[i0], 1);
    disp_entry[i0 * CAP + s0] = t * 2 + 0;
    disp_gate [i0 * CAP + s0] = g0;
    int s1 = atomicAdd(&counts[i1], 1);
    disp_entry[i1 * CAP + s1] = t * 2 + 1;
    disp_gate [i1 * CAP + s1] = g1;
    atomicAdd(&importance[i0], g0);
    atomicAdd(&importance[i1], g1);
  }
}

// ---------------------------------------------------------------- expert FFN (WMMA)
// Block = 8 wave32s; tile = 32 tokens (two 16-row WMMA subtiles sharing B frags).
// K loops use bounded unroll (4) so A fragments stay in registers (no scratch spill).
__global__ __launch_bounds__(256) void ffn_kernel(
    const float* __restrict__ x,
    const int*   __restrict__ disp_entry,
    const float* __restrict__ disp_gate,
    const int*   __restrict__ counts,
    const _Float16* __restrict__ W1T,   // [NE][D_HID][D_IN]
    const float* __restrict__ b1,       // [NE][D_HID]
    const _Float16* __restrict__ W2T,   // [NE][D_IN][D_HID]
    const float* __restrict__ b2,       // [NE][D_IN]
    float* __restrict__ y_pair)         // [2T][D_IN], one writer per element
{
  __shared__ __align__(32) _Float16 Xs[TILE_M][D_IN];    //  64 KB
  __shared__ __align__(32) _Float16 Hs[TILE_M][D_HID];   // 128 KB
  __shared__ int   s_entry[TILE_M];
  __shared__ float s_gate[TILE_M];

  const int e    = blockIdx.y;
  const int cnt  = counts[e];
  const int row0 = blockIdx.x * TILE_M;
  if (row0 >= cnt) return;

  const int tid  = threadIdx.x;
  const int wave = tid >> 5;
  const int lane = tid & 31;

  if (tid < TILE_M) {
    int rr = row0 + tid;
    int ent = 0; float g = 0.f;
    if (rr < cnt) { ent = disp_entry[e * CAP + rr]; g = disp_gate[e * CAP + rr]; }
    s_entry[tid] = ent;
    s_gate[tid]  = g;
  }
  __syncthreads();

  // stage gathered X rows, f32 -> f16
  for (int i = tid; i < TILE_M * D_IN; i += 256) {
    int r = i >> 10, c = i & (D_IN - 1);
    int tok = s_entry[r] >> 1;
    Xs[r][c] = (_Float16)x[(size_t)tok * D_IN + c];
  }
  __syncthreads();

  const int n  = lane & 15;    // N column; also the A-fragment M row for this lane
  const int mh = lane >> 4;    // lane-half selector for K sub-ranges
  const _Float16* W1e = W1T + (size_t)e * D_HID * D_IN;
  const _Float16* W2e = W2T + (size_t)e * D_IN * D_HID;

  // ---- GEMM1: H = relu(X[32,1024] @ W1[1024,2048] + b1) -> Hs (f16)
  for (int nt = wave; nt < D_HID / 16; nt += 8) {
    const int n0 = nt * 16;
    v8f acc0 = {}, acc1 = {};
    const _Float16* Brow = W1e + (size_t)(n0 + n) * D_IN + mh * 16;
#pragma unroll 4
    for (int k0 = 0; k0 < D_IN; k0 += 32) {
      union { v16h v; v8h p[2]; } a0, a1;
      a0.p[0] = *(const v8h*)&Xs[n     ][k0 + mh * 16];
      a0.p[1] = *(const v8h*)&Xs[n     ][k0 + mh * 16 + 8];
      a1.p[0] = *(const v8h*)&Xs[n + 16][k0 + mh * 16];
      a1.p[1] = *(const v8h*)&Xs[n + 16][k0 + mh * 16 + 8];
      v16h b = *(const v16h*)(Brow + k0);     // loaded once, feeds 2 WMMAs
      acc0 = __builtin_amdgcn_wmma_f32_16x16x32_f16(
          false, a0.v, false, b, (short)0, acc0, false, false);
      acc1 = __builtin_amdgcn_wmma_f32_16x16x32_f16(
          false, a1.v, false, b, (short)0, acc1, false, false);
    }
    float bias = b1[e * D_HID + n0 + n];
#pragma unroll
    for (int v = 0; v < 8; ++v) {             // C/D layout: M = v + 8*mh, N = n
      float h0 = acc0[v] + bias;
      float h1 = acc1[v] + bias;
      Hs[v + 8 * mh     ][n0 + n] = (_Float16)(h0 > 0.f ? h0 : 0.f);
      Hs[v + 8 * mh + 16][n0 + n] = (_Float16)(h1 > 0.f ? h1 : 0.f);
    }
  }
  __syncthreads();

  // ---- GEMM2: Y = H[32,2048] @ W2[2048,1024] + b2, scaled by gate, scattered
  for (int nt = wave; nt < D_IN / 16; nt += 8) {
    const int n0 = nt * 16;
    v8f acc0 = {}, acc1 = {};
    const _Float16* Brow = W2e + (size_t)(n0 + n) * D_HID + mh * 16;
#pragma unroll 4
    for (int k0 = 0; k0 < D_HID; k0 += 32) {
      union { v16h v; v8h p[2]; } a0, a1;
      a0.p[0] = *(const v8h*)&Hs[n     ][k0 + mh * 16];
      a0.p[1] = *(const v8h*)&Hs[n     ][k0 + mh * 16 + 8];
      a1.p[0] = *(const v8h*)&Hs[n + 16][k0 + mh * 16];
      a1.p[1] = *(const v8h*)&Hs[n + 16][k0 + mh * 16 + 8];
      v16h b = *(const v16h*)(Brow + k0);     // loaded once, feeds 2 WMMAs
      acc0 = __builtin_amdgcn_wmma_f32_16x16x32_f16(
          false, a0.v, false, b, (short)0, acc0, false, false);
      acc1 = __builtin_amdgcn_wmma_f32_16x16x32_f16(
          false, a1.v, false, b, (short)0, acc1, false, false);
    }
    float bias = b2[e * D_IN + n0 + n];
#pragma unroll
    for (int v = 0; v < 8; ++v) {
      int m0 = v + 8 * mh;
      int m1 = m0 + 16;
      if (row0 + m0 < cnt)
        y_pair[(size_t)s_entry[m0] * D_IN + n0 + n] = s_gate[m0] * (acc0[v] + bias);
      if (row0 + m1 < cnt)
        y_pair[(size_t)s_entry[m1] * D_IN + n0 + n] = s_gate[m1] * (acc1[v] + bias);
    }
  }
}

// ---------------------------------------------------------------- loss (cv^2 + cv^2)
__global__ void loss_kernel(const float* __restrict__ importance,
                            const int* __restrict__ counts,
                            float* __restrict__ out_loss) {
  float mi = 0.f, ml = 0.f;
#pragma unroll
  for (int e = 0; e < NE; ++e) { mi += importance[e]; ml += (float)counts[e]; }
  mi *= (1.f / NE); ml *= (1.f / NE);
  float vi = 0.f, vl = 0.f;
#pragma unroll
  for (int e = 0; e < NE; ++e) {
    float di = importance[e] - mi; vi += di * di;
    float dl = (float)counts[e] - ml; vl += dl * dl;
  }
  vi *= (1.f / NE); vl *= (1.f / NE);
  *out_loss = vi / (mi * mi + 1e-10f) + vl / (ml * ml + 1e-10f);
}

// ---------------------------------------------------------------- residual + LayerNorm
__global__ __launch_bounds__(256) void ln_kernel(
    const float* __restrict__ x, const float* __restrict__ y_pair,
    const float* __restrict__ gamma, const float* __restrict__ beta,
    float* __restrict__ out) {
  __shared__ float sh_s[8], sh_q[8];
  const int t = blockIdx.x, tid = threadIdx.x;
  const float* xr = x + (size_t)t * D_IN;
  const float* p0 = y_pair + (size_t)(2 * t) * D_IN;
  const float* p1 = p0 + D_IN;

  float vals[4], s = 0.f, q = 0.f;
#pragma unroll
  for (int j = 0; j < 4; ++j) {
    int d = tid + j * 256;
    float v = xr[d] + p0[d] + p1[d];
    vals[j] = v; s += v; q += v * v;
  }
#pragma unroll
  for (int off = 16; off; off >>= 1) {
    s += __shfl_down(s, off, 32);
    q += __shfl_down(q, off, 32);
  }
  int wave = tid >> 5, lane = tid & 31;
  if (lane == 0) { sh_s[wave] = s; sh_q[wave] = q; }
  __syncthreads();
  if (tid == 0) {
    float S = 0.f, Q = 0.f;
#pragma unroll
    for (int w = 0; w < 8; ++w) { S += sh_s[w]; Q += sh_q[w]; }
    sh_s[0] = S; sh_q[0] = Q;
  }
  __syncthreads();
  float mu  = sh_s[0] * (1.f / D_IN);
  float var = sh_q[0] * (1.f / D_IN) - mu * mu;
  float inv = rsqrtf(var + 1e-5f);
  float* outr = out + (size_t)t * D_IN;
#pragma unroll
  for (int j = 0; j < 4; ++j) {
    int d = tid + j * 256;
    outr[d] = (vals[j] - mu) * inv * gamma[d] + beta[d];
  }
}

// ---------------------------------------------------------------- launcher
extern "C" void kernel_launch(void* const* d_in, const int* in_sizes, int n_in,
                              void* d_out, int out_size, void* d_ws, size_t ws_size,
                              hipStream_t stream) {
  const float* x     = (const float*)d_in[0];   // [4,2048,1024]
  const float* Wg    = (const float*)d_in[1];   // [1024,8]
  const float* W1    = (const float*)d_in[2];   // [8,1024,2048]
  const float* b1    = (const float*)d_in[3];   // [8,2048]
  const float* W2    = (const float*)d_in[4];   // [8,2048,1024]
  const float* b2    = (const float*)d_in[5];   // [8,1024]
  const float* gamma = (const float*)d_in[6];   // [1024]
  const float* beta  = (const float*)d_in[7];   // [1024]
  float* out = (float*)d_out;                   // [T*D_IN] y  + [1] loss

  // workspace carve-up (256B aligned)
  char* ws = (char*)d_ws;
  size_t off = 0;
  auto take = [&](size_t bytes) -> char* {
    char* p = ws + off;
    off += (bytes + 255) & ~(size_t)255;
    return p;
  };
  const size_t wElems = (size_t)NE * D_IN * D_HID;            // 16,777,216
  _Float16* W1T       = (_Float16*)take(wElems * sizeof(_Float16));
  _Float16* W2T       = (_Float16*)take(wElems * sizeof(_Float16));
  float*    y_pair    = (float*)take((size_t)2 * T_TOK * D_IN * sizeof(float));
  int*      disp_ent  = (int*)take((size_t)NE * CAP * sizeof(int));
  float*    disp_gate = (float*)take((size_t)NE * CAP * sizeof(float));
  int*      counts    = (int*)take(NE * sizeof(int));
  float*    importance= (float*)take(NE * sizeof(float));

  zero_kernel<<<1, 64, 0, stream>>>(counts, importance);
  wprep_kernel<<<(int)(wElems / 256), 256, 0, stream>>>(W1, W2, W1T, W2T);
  gate_kernel<<<T_TOK / 8, 256, 0, stream>>>(x, Wg, counts, importance,
                                             disp_ent, disp_gate);
  ffn_kernel<<<dim3(MAX_TILES, NE), 256, 0, stream>>>(
      x, disp_ent, disp_gate, counts, W1T, b1, W2T, b2, y_pair);
  loss_kernel<<<1, 1, 0, stream>>>(importance, counts,
                                   out + (size_t)T_TOK * D_IN);
  ln_kernel<<<T_TOK, 256, 0, stream>>>(x, y_pair, gamma, beta, out);
}
